// THGNNModel_33500744909283
// MI455X (gfx1250) — matrix operand order repaired and checked by reading, not compiled, not run on hardware
//
#include <hip/hip_runtime.h>
#include <hip/hip_bf16.h>
#include <math.h>

typedef __attribute__((ext_vector_type(16))) _Float16 v16h;
typedef __attribute__((ext_vector_type(2)))  _Float16 v2h;
typedef __attribute__((ext_vector_type(8)))  float    v8f;

#define GEMM_BM 64
#define GEMM_BN 64
#define GEMM_KT 32
#define FRAG_PAD 24   // halves per fragment row in LDS (16 used + pad, 48B keeps 16B align)

// ---------------------------------------------------------------------------
// Generic WMMA GEMM:  C = act(A @ B + bias) (+ resid)
//   A: M x K (f32, row-major)
//   BMODE 0: B[k][n] = W[n*K + k]            (weight stored [N,K], compute A@W^T)
//   BMODE 1: B[k][n] = W[k*N + n]            (plain K x N matrix, compute A@B)
//   BMODE 2: B[k][n] = W[n*384 + (k%128)*3 + (k/128)]   (conv1d weight [O,I,3], K=384)
//   FULL: M%64==0 && N%64==0 && K%32==0 -> no bounds checks, b128 global loads.
//   act 0: none, 1: relu.  resid (nullable): C += resid (shape M x N)
//   batching over blockIdx.z; B offset uses (z % modB) when modB>0.
//
// LDS is kept in *fragment-native* layout so each lane reads its WMMA operand
// as one contiguous 32-byte load (2x ds_load_b128).
//   A frag (ISA 7.12.2): lane l row = l%16, elem e -> K = e + (e>=8?8:0) + (l>=16?8:0)
//   B frag:              lane l col = l%16, elem e -> K = e + (e>=8?8:0) + (l>=16?8:0)
// ---------------------------------------------------------------------------
template <int BMODE, bool FULL>
__global__ __launch_bounds__(128) void gemm_wmma(
    const float* __restrict__ A, const float* __restrict__ B,
    const float* __restrict__ bias, const float* __restrict__ resid,
    float* __restrict__ C, int M, int N, int K,
    long long strideA, long long strideB, int modB,
    long long strideC, long long strideBias, int act)
{
    const int z = blockIdx.z;
    A += (long long)z * strideA;
    B += (long long)(modB ? (z % modB) : z) * strideB;
    C += (long long)z * strideC;
    const float* biasP  = bias  ? bias  + (long long)z * strideBias : nullptr;
    const float* residP = resid ? resid + (long long)z * strideC   : nullptr;

    __shared__ __align__(16) _Float16 sAf[4][32][FRAG_PAD];  // [mtile][lane][e]
    __shared__ __align__(16) _Float16 sBf[4][32][FRAG_PAD];  // [ntile][lane][e]

    const int tid  = threadIdx.x;
    const int lane = tid & 31;
    const int wv   = tid >> 5;           // 4 waves, each owns a 16-row strip
    const int m0   = blockIdx.y * GEMM_BM;
    const int n0   = blockIdx.x * GEMM_BN;

    v8f acc[4] = {};

    for (int k0 = 0; k0 < K; k0 += GEMM_KT) {
        if (FULL) {
            // ---- unguarded staging: 512 K-quads per tile, 4 per thread ----
            #pragma unroll
            for (int t = 0; t < 4; ++t) {
                const int p = tid + t * 128;
                const int r = p >> 3;              // 0..63 tile row
                const int c = (p & 7) << 2;        // 0,4,..,28
                const float* ap = A + (long long)(m0 + r) * K + (k0 + c);
                const float4 v = *(const float4*)ap;
                if (t == 0 && k0 + GEMM_KT < K) __builtin_prefetch(ap + GEMM_KT, 0, 3);
                auto pk0 = __builtin_amdgcn_cvt_pkrtz(v.x, v.y);   // v_cvt_pk_rtz_f16_f32
                auto pk1 = __builtin_amdgcn_cvt_pkrtz(v.z, v.w);
                const int mt = r >> 4;
                const int l  = (r & 15) + (((c >> 3) & 1) << 4);
                const int e  = (c & 7) + ((c >> 4) << 3);
                *(decltype(pk0)*)&sAf[mt][l][e]     = pk0;
                *(decltype(pk1)*)&sAf[mt][l][e + 2] = pk1;
            }
            #pragma unroll
            for (int t = 0; t < 4; ++t) {
                const int p = tid + t * 128;
                const int n  = p >> 3;             // 0..63 tile col
                const int kk = (p & 7) << 2;       // 0,4,..,28
                const int gn = n0 + n;
                const int gk = k0 + kk;
                float x0, x1, x2, x3;
                if (BMODE == 0) {
                    const float* bp = B + (long long)gn * K + gk;
                    const float4 v = *(const float4*)bp;
                    if (t == 0 && k0 + GEMM_KT < K) __builtin_prefetch(bp + GEMM_KT, 0, 3);
                    x0 = v.x; x1 = v.y; x2 = v.z; x3 = v.w;
                } else if (BMODE == 1) {
                    x0 = B[(long long)gk * N + gn];
                    x1 = B[(long long)(gk + 1) * N + gn];
                    x2 = B[(long long)(gk + 2) * N + gn];
                    x3 = B[(long long)(gk + 3) * N + gn];
                } else {
                    const float* bp = B + (long long)gn * 384 + (long long)(gk & 127) * 3 + (gk >> 7);
                    x0 = bp[0]; x1 = bp[3]; x2 = bp[6]; x3 = bp[9];
                }
                auto pk0 = __builtin_amdgcn_cvt_pkrtz(x0, x1);
                auto pk1 = __builtin_amdgcn_cvt_pkrtz(x2, x3);
                const int nt = n >> 4;
                const int l  = (n & 15) + (((kk >> 3) & 1) << 4);
                const int e  = (kk & 7) + ((kk >> 4) << 3);
                *(decltype(pk0)*)&sBf[nt][l][e]     = pk0;
                *(decltype(pk1)*)&sBf[nt][l][e + 2] = pk1;
            }
        } else {
            // ---- guarded staging: 1024 K-pairs per tile, 8 per thread ----
            #pragma unroll
            for (int t = 0; t < 8; ++t) {
                const int p = tid + t * 128;
                const int r = p >> 4;              // 0..63 tile row
                const int c = (p & 15) << 1;       // 0,2,..,30
                const int gr = m0 + r;
                const int gk = k0 + c;
                float v0 = 0.f, v1 = 0.f;
                if (gr < M) {
                    const float* ap = A + (long long)gr * K + gk;
                    if (gk < K)     v0 = ap[0];
                    if (gk + 1 < K) v1 = ap[1];
                }
                auto pk = __builtin_amdgcn_cvt_pkrtz(v0, v1);
                const int mt = r >> 4;
                const int l  = (r & 15) + (((c >> 3) & 1) << 4);
                const int e  = (c & 7) + ((c >> 4) << 3);
                *(decltype(pk)*)&sAf[mt][l][e] = pk;
            }
            #pragma unroll
            for (int t = 0; t < 8; ++t) {
                const int p = tid + t * 128;
                const int n  = p >> 4;
                const int kk = (p & 15) << 1;
                const int gn = n0 + n;
                const int gk = k0 + kk;
                float v0 = 0.f, v1 = 0.f;
                if (gn < N) {
                    if (BMODE == 0) {
                        const float* bp = B + (long long)gn * K + gk;
                        if (gk < K)     v0 = bp[0];
                        if (gk + 1 < K) v1 = bp[1];
                    } else if (BMODE == 1) {
                        if (gk < K)     v0 = B[(long long)gk * N + gn];
                        if (gk + 1 < K) v1 = B[(long long)(gk + 1) * N + gn];
                    } else {
                        if (gk < K)     v0 = B[(long long)gn * 384 + (long long)(gk & 127) * 3 + (gk >> 7)];
                        if (gk + 1 < K) v1 = B[(long long)gn * 384 + (long long)((gk + 1) & 127) * 3 + ((gk + 1) >> 7)];
                    }
                }
                auto pk = __builtin_amdgcn_cvt_pkrtz(v0, v1);
                const int nt = n >> 4;
                const int l  = (n & 15) + (((kk >> 3) & 1) << 4);
                const int e  = (kk & 7) + ((kk >> 4) << 3);
                *(decltype(pk)*)&sBf[nt][l][e] = pk;
            }
        }
        __syncthreads();

        // ---- compute: fragments are contiguous per lane -> wide DS loads ----
        const v16h af = *(const v16h*)&sAf[wv][lane][0];
        #pragma unroll
        for (int nt = 0; nt < 4; ++nt) {
            const v16h bf = *(const v16h*)&sBf[nt][lane][0];
            acc[nt] = __builtin_amdgcn_wmma_f32_16x16x32_f16(
                false, af, false, bf, (short)0, acc[nt], false, false);
        }
        __syncthreads();
    }

    // C layout: VGPR i -> row (i | i+8 for upper lane half), col = lane%16
    const int rowBase = m0 + (wv << 4) + ((lane >> 4) << 3);
    #pragma unroll
    for (int nt = 0; nt < 4; ++nt) {
        int gn = n0 + (nt << 4) + (lane & 15);
        if (!FULL && gn >= N) continue;
        float bv = biasP ? biasP[gn] : 0.f;
        #pragma unroll
        for (int i = 0; i < 8; ++i) {
            int gm = rowBase + i;
            if (!FULL && gm >= M) continue;
            float v = acc[nt][i] + bv;
            if (act == 1) v = fmaxf(v, 0.f);
            long long off = (long long)gm * N + gn;
            if (residP) v += residP[off];
            C[off] = v;
        }
    }
}

// ---------------------------------------------------------------------------
// Fused softmax attention for one (sequence, head). qkv rows: [seq*sLen + s] x 384
// (q cols 0..127, k 128..255, v 256..383; head block = hd*dHead). out: rows x 128.
// ---------------------------------------------------------------------------
template <int SMAX, int DMAX>
__global__ __launch_bounds__(128) void attn_kernel(
    const float* __restrict__ qkv, float* __restrict__ out,
    int sLen, int dHead, float scale)
{
    __shared__ float qs[SMAX * DMAX];
    __shared__ float ks[SMAX * DMAX];
    __shared__ float vs[SMAX * DMAX];
    __shared__ float sc[SMAX * SMAX];

    const int seq = blockIdx.x, hd = blockIdx.y, tid = threadIdx.x;
    const int tot = sLen * dHead;
    for (int i = tid; i < tot; i += 128) {
        int r = i / dHead, c = i - r * dHead;
        long long base = ((long long)seq * sLen + r) * 384 + hd * dHead + c;
        qs[i] = qkv[base];
        ks[i] = qkv[base + 128];
        vs[i] = qkv[base + 256];
    }
    __syncthreads();

    if (tid < sLen) {
        const int q = tid;
        float mx = -1e30f;
        for (int j = 0; j < sLen; ++j) {
            float s = 0.f;
            for (int d = 0; d < dHead; ++d) s += qs[q * dHead + d] * ks[j * dHead + d];
            s *= scale;
            sc[q * sLen + j] = s;
            mx = fmaxf(mx, s);
        }
        float sum = 0.f;
        for (int j = 0; j < sLen; ++j) {
            float e = expf(sc[q * sLen + j] - mx);
            sc[q * sLen + j] = e;
            sum += e;
        }
        float inv = 1.f / sum;
        for (int d = 0; d < dHead; ++d) {
            float o = 0.f;
            for (int j = 0; j < sLen; ++j) o += sc[q * sLen + j] * vs[j * dHead + d];
            out[((long long)seq * sLen + q) * 128 + hd * dHead + d] = o * inv;
        }
    }
}

// ---------------------------------------------------------------------------
// Small elementwise / data-movement kernels
// ---------------------------------------------------------------------------
__global__ void k_add_pos(const float* __restrict__ x, const float* __restrict__ pos,
                          float* __restrict__ h, long long n)
{
    long long i = (long long)blockIdx.x * blockDim.x + threadIdx.x;
    if (i >= n) return;
    int col = (int)(i & 127);
    long long row = i >> 7;
    int t = (int)(row % 100);
    h[i] = x[i] + pos[t * 128 + col];
}

__global__ void k_im2col(const float* __restrict__ a, float* __restrict__ col, long long n)
{
    long long i = (long long)blockIdx.x * blockDim.x + threadIdx.x;
    if (i >= n) return;
    int c = (int)(i % 384);
    long long row = i / 384;
    int t = c >> 7, ic = c & 127;
    int s = (int)(row % 100);
    long long seq = row / 100;
    int ss = s + t - 1;
    float v = 0.f;
    if (ss >= 0 && ss < 100) v = a[(seq * 100 + ss) * 128 + ic];
    col[i] = v;
}

__global__ void k_gather_last(const float* __restrict__ x, float* __restrict__ dst, long long n)
{
    long long i = (long long)blockIdx.x * blockDim.x + threadIdx.x;
    if (i >= n) return;
    int col = (int)(i & 127);
    long long bm = i >> 7;
    long long bb = bm / 400, nn = bm % 400;
    dst[i] = x[(((bb * 100) + 99) * 400 + nn) * 128 + col];
}

__global__ void k_trans_ai(const float* __restrict__ tm, float* __restrict__ ai, long long n)
{
    long long i = (long long)blockIdx.x * blockDim.x + threadIdx.x;
    if (i >= n) return;
    int col = (int)(i & 127);
    long long row = i >> 7;
    int r = (int)(row & 7);
    long long bm = row >> 3;
    ai[i] = tm[((long long)r * 800 + bm) * 128 + col];
}

// mean over R, residual add, LayerNorm, scatter into x[:, -1]
__global__ __launch_bounds__(128) void k_graph_finish(
    const float* __restrict__ gout, const float* __restrict__ nf,
    const float* __restrict__ g, const float* __restrict__ bln,
    float* __restrict__ x)
{
    const int bm = blockIdx.x, h = threadIdx.x;
    float fm = 0.f;
    #pragma unroll
    for (int r = 0; r < 8; ++r) fm += gout[((long long)bm * 8 + r) * 128 + h];
    fm *= 0.125f;
    float v = nf[(long long)bm * 128 + h] + fm;

    __shared__ float red[128];
    red[h] = v; __syncthreads();
    for (int st = 64; st > 0; st >>= 1) { if (h < st) red[h] += red[h + st]; __syncthreads(); }
    float mu = red[0] * (1.f / 128.f);
    __syncthreads();
    float d = v - mu;
    red[h] = d * d; __syncthreads();
    for (int st = 64; st > 0; st >>= 1) { if (h < st) red[h] += red[h + st]; __syncthreads(); }
    float var = red[0] * (1.f / 128.f);
    float o = d * rsqrtf(var + 1e-5f) * g[h] + bln[h];
    int bb = bm / 400, nn = bm % 400;
    x[(((long long)bb * 100) + 99) * 400 * 128 + (long long)nn * 128 + h] = o;
}

__global__ void k_tanh(float* p, long long n) {
    long long i = (long long)blockIdx.x * blockDim.x + threadIdx.x;
    if (i < n) p[i] = tanhf(p[i]);
}
__global__ void k_softplus(float* p, long long n) {
    long long i = (long long)blockIdx.x * blockDim.x + threadIdx.x;
    if (i < n) { float x = p[i]; p[i] = fmaxf(x, 0.f) + log1pf(expf(-fabsf(x))); }
}
__global__ void k_sigmoid(float* p, long long n) {
    long long i = (long long)blockIdx.x * blockDim.x + threadIdx.x;
    if (i < n) p[i] = 1.f / (1.f + expf(-p[i]));
}

// reg head: mean over nodes -> 128->32 relu -> 32->4 -> softmax, per batch
__global__ __launch_bounds__(128) void k_reg_head(
    const float* __restrict__ fin, const float* __restrict__ w1, const float* __restrict__ b1,
    const float* __restrict__ w2, const float* __restrict__ b2, float* __restrict__ out)
{
    const int bb = blockIdx.x, tid = threadIdx.x;
    __shared__ float mean[128];
    __shared__ float hid[32];
    __shared__ float logits[4];
    float acc = 0.f;
    for (int nn = 0; nn < 400; ++nn) acc += fin[((long long)bb * 400 + nn) * 128 + tid];
    mean[tid] = acc * (1.f / 400.f);
    __syncthreads();
    if (tid < 32) {
        float s = b1[tid];
        for (int k = 0; k < 128; ++k) s += mean[k] * w1[tid * 128 + k];
        hid[tid] = fmaxf(s, 0.f);
    }
    __syncthreads();
    if (tid < 4) {
        float s = b2[tid];
        for (int k = 0; k < 32; ++k) s += hid[k] * w2[tid * 32 + k];
        logits[tid] = s;
    }
    __syncthreads();
    if (tid == 0) {
        float mx = fmaxf(fmaxf(logits[0], logits[1]), fmaxf(logits[2], logits[3]));
        float e0 = expf(logits[0] - mx), e1 = expf(logits[1] - mx);
        float e2 = expf(logits[2] - mx), e3 = expf(logits[3] - mx);
        float s = e0 + e1 + e2 + e3;
        out[bb * 4 + 0] = e0 / s; out[bb * 4 + 1] = e1 / s;
        out[bb * 4 + 2] = e2 / s; out[bb * 4 + 3] = e3 / s;
    }
}

// ---------------------------------------------------------------------------
// Host-side orchestration
// ---------------------------------------------------------------------------
static inline dim3 grid1(long long n, int bs = 256) { return dim3((unsigned)((n + bs - 1) / bs)); }

extern "C" void kernel_launch(void* const* d_in, const int* in_sizes, int n_in,
                              void* d_out, int out_size, void* d_ws, size_t ws_size,
                              hipStream_t stream)
{
    (void)in_sizes; (void)n_in; (void)out_size; (void)ws_size;
    const float* td      = (const float*)d_in[0];
    const float* adj     = (const float*)d_in[1];
    const float* in_w    = (const float*)d_in[2];
    const float* in_b    = (const float*)d_in[3];
    const float* pos     = (const float*)d_in[4];
    const float* t_qkv_w = (const float*)d_in[5];
    const float* t_qkv_b = (const float*)d_in[6];
    const float* t_out_w = (const float*)d_in[7];
    const float* t_out_b = (const float*)d_in[8];
    const float* conv_w  = (const float*)d_in[9];
    const float* conv_b  = (const float*)d_in[10];
    const float* rel_w   = (const float*)d_in[11];
    const float* rel_b   = (const float*)d_in[12];
    const float* g_qkv_w = (const float*)d_in[13];
    const float* g_qkv_b = (const float*)d_in[14];
    const float* g_out_w = (const float*)d_in[15];
    const float* g_out_b = (const float*)d_in[16];
    const float* ln_g    = (const float*)d_in[17];
    const float* ln_b    = (const float*)d_in[18];
    const float* pm_w1   = (const float*)d_in[19];
    const float* pm_b1   = (const float*)d_in[20];
    const float* pm_w2   = (const float*)d_in[21];
    const float* pm_b2   = (const float*)d_in[22];
    const float* vol_w1  = (const float*)d_in[23];
    const float* vol_b1  = (const float*)d_in[24];
    const float* vol_w2  = (const float*)d_in[25];
    const float* vol_b2  = (const float*)d_in[26];
    const float* reg_w1  = (const float*)d_in[27];
    const float* reg_b1  = (const float*)d_in[28];
    const float* reg_w2  = (const float*)d_in[29];
    const float* reg_b2  = (const float*)d_in[30];
    const float* conf_w1 = (const float*)d_in[31];
    const float* conf_b1 = (const float*)d_in[32];
    const float* conf_w2 = (const float*)d_in[33];
    const float* conf_b2 = (const float*)d_in[34];

    float* ws  = (float*)d_ws;
    float* out = (float*)d_out;

    // workspace layout (floats)
    const long long ROWS = 80000;                 // B*S*N
    float* x    = ws;                             // 10,240,000  (b,s,n,128)
    float* buf2 = ws + 10240000;                  // h then o    (80000 x 128)
    float* aBuf = ws + 20480000;                  // a           (80000 x 128)
    float* qbuf = ws + 30720000;                  // qkv / im2col (80000 x 384)
    // graph temps reuse the qkv region (free by the time graph phase runs)
    float* nf   = qbuf;                           // 800 x 128
    float* msgs = nf   + 102400;                  // 8 x 800 x 128
    float* tmB  = msgs + 819200;                  // 8 x 800 x 128
    float* ai   = tmB  + 819200;                  // 6400 x 128
    float* gqkv = ai   + 819200;                  // 6400 x 384
    float* gatt = gqkv + 2457600;                 // 6400 x 128
    float* gout = gatt + 819200;                  // 6400 x 128
    float* h1   = gout + 819200;                  // 800 x 64 scratch (heads)

    float* finalBuf = out + 8808;                 // 'final' output slot (800 x 128)

    auto gemm = [&](const float* A, const float* B, const float* bias, const float* resid,
                    float* C, int M, int N, int K, int batch,
                    long long sA, long long sB, int modB, long long sC, long long sBias,
                    int bmode, int act) {
        dim3 g((N + GEMM_BN - 1) / GEMM_BN, (M + GEMM_BM - 1) / GEMM_BM, batch);
        const bool full = (M % GEMM_BM == 0) && (N % GEMM_BN == 0) && (K % GEMM_KT == 0);
        if (bmode == 0) {
            if (full) gemm_wmma<0, true ><<<g, 128, 0, stream>>>(A, B, bias, resid, C, M, N, K, sA, sB, modB, sC, sBias, act);
            else      gemm_wmma<0, false><<<g, 128, 0, stream>>>(A, B, bias, resid, C, M, N, K, sA, sB, modB, sC, sBias, act);
        } else if (bmode == 1) {
            if (full) gemm_wmma<1, true ><<<g, 128, 0, stream>>>(A, B, bias, resid, C, M, N, K, sA, sB, modB, sC, sBias, act);
            else      gemm_wmma<1, false><<<g, 128, 0, stream>>>(A, B, bias, resid, C, M, N, K, sA, sB, modB, sC, sBias, act);
        } else {
            if (full) gemm_wmma<2, true ><<<g, 128, 0, stream>>>(A, B, bias, resid, C, M, N, K, sA, sB, modB, sC, sBias, act);
            else      gemm_wmma<2, false><<<g, 128, 0, stream>>>(A, B, bias, resid, C, M, N, K, sA, sB, modB, sC, sBias, act);
        }
    };

    // input projection: x = td @ in_w^T + in_b   (K=20 zero-padded in LDS)
    gemm(td, in_w, in_b, nullptr, x, (int)ROWS, 128, 20, 1, 0, 0, 0, 0, 0, 0, 0);

    for (int l = 0; l < 4; ++l) {
        // temporal layer
        k_add_pos<<<grid1(10240000), 256, 0, stream>>>(x, pos + (long long)l * 12800, buf2, 10240000);
        gemm(buf2, t_qkv_w + (long long)l * 49152, t_qkv_b + l * 384, nullptr, qbuf,
             (int)ROWS, 384, 128, 1, 0, 0, 0, 0, 0, 0, 0);
        attn_kernel<100, 16><<<dim3(800, 8), 128, 0, stream>>>(qbuf, buf2, 100, 16, 0.25f);
        gemm(buf2, t_out_w + (long long)l * 16384, t_out_b + l * 128, nullptr, aBuf,
             (int)ROWS, 128, 128, 1, 0, 0, 0, 0, 0, 0, 0);
        k_im2col<<<grid1(30720000), 256, 0, stream>>>(aBuf, qbuf, 30720000);
        // x = a + conv(a): conv as GEMM with conv-weight layout, fused residual
        gemm(qbuf, conv_w + (long long)l * 49152, conv_b + l * 128, aBuf, x,
             (int)ROWS, 128, 384, 1, 0, 0, 0, 0, 0, 2, 0);

        // graph layer on x[:, -1]
        k_gather_last<<<grid1(102400), 256, 0, stream>>>(x, nf, 102400);
        // msgs[r,b] = adj[r,b] @ nf[b]   (batch z = r*2+b; B offset uses z%2)
        gemm(adj, nf, nullptr, nullptr, msgs, 400, 128, 400, 16,
             160000, 51200, 2, 51200, 0, 1, 0);
        // tm[r] = msgs[r] @ rel_w[l,r]^T + rel_b[l,r]
        gemm(msgs, rel_w + (long long)l * 131072, rel_b + (long long)l * 1024, nullptr, tmB,
             800, 128, 128, 8, 102400, 16384, 0, 102400, 128, 0, 0);
        k_trans_ai<<<grid1(819200), 256, 0, stream>>>(tmB, ai, 819200);
        gemm(ai, g_qkv_w + (long long)l * 49152, g_qkv_b + l * 384, nullptr, gqkv,
             6400, 384, 128, 1, 0, 0, 0, 0, 0, 0, 0);
        attn_kernel<8, 32><<<dim3(800, 4), 128, 0, stream>>>(gqkv, gatt, 8, 32, 0.17677669529663687f);
        gemm(gatt, g_out_w + (long long)l * 16384, g_out_b + l * 128, nullptr, gout,
             6400, 128, 128, 1, 0, 0, 0, 0, 0, 0, 0);
        k_graph_finish<<<800, 128, 0, stream>>>(gout, nf, ln_g + l * 128, ln_b + l * 128, x);
    }

    // final = x[:, -1]  -> directly into its output slot
    k_gather_last<<<grid1(102400), 256, 0, stream>>>(x, finalBuf, 102400);

    // pm = tanh(relu(final@w1^T+b1)@w2^T+b2)   -> out[0 .. 4000)
    gemm(finalBuf, pm_w1, pm_b1, nullptr, h1, 800, 64, 128, 1, 0, 0, 0, 0, 0, 0, 1);
    gemm(h1, pm_w2, pm_b2, nullptr, out, 800, 5, 64, 1, 0, 0, 0, 0, 0, 0, 0);
    k_tanh<<<grid1(4000), 256, 0, stream>>>(out, 4000);

    // vol = softplus(...)                      -> out[4000 .. 8000)
    gemm(finalBuf, vol_w1, vol_b1, nullptr, h1, 800, 64, 128, 1, 0, 0, 0, 0, 0, 0, 1);
    gemm(h1, vol_w2, vol_b2, nullptr, out + 4000, 800, 5, 64, 1, 0, 0, 0, 0, 0, 0, 0);
    k_softplus<<<grid1(4000), 256, 0, stream>>>(out + 4000, 4000);

    // reg = softmax(head2(mean(final)))        -> out[8000 .. 8008)
    k_reg_head<<<2, 128, 0, stream>>>(finalBuf, reg_w1, reg_b1, reg_w2, reg_b2, out + 8000);

    // conf = sigmoid(...)                      -> out[8008 .. 8808)
    gemm(finalBuf, conf_w1, conf_b1, nullptr, h1, 800, 32, 128, 1, 0, 0, 0, 0, 0, 0, 1);
    gemm(h1, conf_w2, conf_b2, nullptr, out + 8008, 800, 1, 32, 1, 0, 0, 0, 0, 0, 0, 0);
    k_sigmoid<<<grid1(800), 256, 0, stream>>>(out + 8008, 800);
}
